// SUBNET_58179626992069
// MI455X (gfx1250) — matrix operand order repaired and checked by reading, not compiled, not run on hardware
//
#include <hip/hip_runtime.h>
#include <hip/hip_bf16.h>

// ---------------------------------------------------------------------------
// Fused subnet rollout for MI455X (gfx1250), wave32 + V_WMMA_F32_16X16X32_BF16
//   - one wave per workgroup owns a 16-row batch tile (512 independent waves)
//   - all rollout weights (60 B-fragments = 480 VGPRs) live in registers
//   - activations in LDS as bf16 row-major -> A-frag = 8 x ds_load_b32
//   - hardware v_tanh_f32; native bf16 converts (v_cvt_pk_bf16_f32)
// ---------------------------------------------------------------------------

#define B_TOT 8192
#define T_    128
#define NX    64
#define NU    16
#define NY    16
#define NH    64
#define ASTR  72   // LDS activation row stride in bf16 elems (36 dwords; 36*m % 64 all-distinct)

typedef __attribute__((ext_vector_type(16))) __bf16       v16bf;
typedef __attribute__((ext_vector_type(2)))  __bf16       v2bf;
typedef __attribute__((ext_vector_type(8)))  float        v8f;
typedef __attribute__((ext_vector_type(8)))  unsigned int v8u;

// packed-weight offsets in d_ws (dwords). One B-fragment = 32 lanes * 8 dwords.
#define OFF_ENC_W1   0        // 1024x64 : 4 nTiles * 32 kChunks * 256
#define OFF_ENC_WRES 32768
#define OFF_ENC_W2   65536    // 64x64   : 4 * 2 * 256
#define OFF_ENC_W3   67584
#define OFF_F_W1     69632    // 80x64 (pad 96) : 4 * 3 * 256
#define OFF_F_WRES   72704
#define OFF_F_W2     75776
#define OFF_F_W3     77824
#define OFF_H_W1     79872
#define OFF_H_W2     81920
#define OFF_H_WRES   83968    // 64x16 : 1 * 2 * 256
#define OFF_H_W3     84480
#define WS_DWORDS    84992    // ~340 KB of d_ws

// -------------------------------- helpers ----------------------------------

// native bf16 converts: clang lowers these to V_CVT_PK_BF16_F32 on gfx1250
__device__ __forceinline__ unsigned int pack2bf(float a, float b) {
  v2bf v;
  v[0] = (__bf16)a;
  v[1] = (__bf16)b;
  return __builtin_bit_cast(unsigned int, v);
}

__device__ __forceinline__ unsigned short f2bf(float f) {
  return __builtin_bit_cast(unsigned short, (__bf16)f);
}

__device__ __forceinline__ float fast_tanh(float x) {
#if __has_builtin(__builtin_amdgcn_tanhf)
  return __builtin_amdgcn_tanhf(x);       // gfx1250 V_TANH_F32
#elif __has_builtin(__builtin_amdgcn_tanh_f32)
  return __builtin_amdgcn_tanh_f32(x);
#else
  return tanhf(x);
#endif
}

__device__ __forceinline__ v8f vzero() {
  v8f v;
#pragma unroll
  for (int i = 0; i < 8; ++i) v[i] = 0.f;
  return v;
}

__device__ __forceinline__ v8f wmma_bf16(v16bf a, v16bf b, v8f c) {
  return __builtin_amdgcn_wmma_f32_16x16x32_bf16(false, a, false, b,
                                                 (short)0, c, false, false);
}

// B fragment load (packed layout produced by pack_w)
__device__ __forceinline__ v16bf b_frag(const unsigned int* __restrict__ Wp,
                                        int n, int c, int nC, int lane) {
  const unsigned int* p = Wp + (((n * nC) + c) * 32 + lane) * 8;
  return *reinterpret_cast<const v16bf*>(p);
}

// A fragment from a bf16 LDS tile [16][ASTR]: pairs (k,k+1) are one dword.
// 16-bit A layout: lane half kb in {0,8}; reg r -> K = k0+kb+(r>=4?16:0)+2*(r&3)
__device__ __forceinline__ v16bf a_ldsh(const unsigned short* __restrict__ buf,
                                        int k0, int lane) {
  int m  = lane & 15;
  int kb = (lane & 16) ? 8 : 0;
  const unsigned int* row =
      reinterpret_cast<const unsigned int*>(buf + m * ASTR);
  v8u t;
#pragma unroll
  for (int r = 0; r < 8; ++r) {
    int k = k0 + kb + ((r & 4) << 2) + 2 * (r & 3);
    t[r] = row[k >> 1];
  }
  return __builtin_bit_cast(v16bf, t);
}

// per-lane normalization constants: mean/inv-std pairs for k = kb + 2j (+1)
struct UN { float2 m[4]; float2 i[4]; };

// D tile (f32 16x16) -> bf16 LDS with scalar bias (+ optional tanh)
template <bool TANH>
__device__ __forceinline__ void d_storeh(unsigned short* __restrict__ buf, int n0,
                                         v8f d, float bv, int lane) {
  int n  = n0 + (lane & 15);
  int mb = (lane & 16) ? 8 : 0;
#pragma unroll
  for (int v = 0; v < 8; ++v) {
    float val = d[v] + bv;
    if (TANH) val = fast_tanh(val);
    buf[(mb + v) * ASTR + n] = f2bf(val);
  }
}

// ------------------------- weight pre-pack kernel --------------------------
// W[K][N] row-major f32 -> bf16 B-fragments: dword idx = ((nT*nC+c)*32+lane)*8+r
__global__ void pack_w(const float* __restrict__ W, unsigned int* __restrict__ dst,
                       int K, int Kp, int N) {
  int nC = Kp >> 5, nT = N >> 4;
  int total = nT * nC * 256;
  int idx = blockIdx.x * blockDim.x + threadIdx.x;
  if (idx >= total) return;
  int r    = idx & 7;
  int lane = (idx >> 3) & 31;
  int g    = idx >> 8;
  int c    = g % nC;
  int n    = (g / nC) * 16 + (lane & 15);
  int k    = c * 32 + ((lane & 16) ? 8 : 0) + ((r & 4) << 2) + 2 * (r & 3);
  float f0 = (k     < K) ? W[k * N + n]       : 0.f;
  float f1 = (k + 1 < K) ? W[(k + 1) * N + n] : 0.f;
  dst[idx] = pack2bf(f0, f1);
}

// ------------------------------ main kernel --------------------------------

__global__ __launch_bounds__(32) void subnet_kernel(
    const float* __restrict__ upast, const float* __restrict__ ypast,
    const float* __restrict__ ufuture,
    const float* __restrict__ enc_bres, const float* __restrict__ enc_b1,
    const float* __restrict__ enc_b2,  const float* __restrict__ enc_b3,
    const float* __restrict__ f_bres,  const float* __restrict__ f_b1,
    const float* __restrict__ f_b2,    const float* __restrict__ f_b3,
    const float* __restrict__ h_bres,  const float* __restrict__ h_b1,
    const float* __restrict__ h_b2,    const float* __restrict__ h_b3,
    const float* __restrict__ u_mean,  const float* __restrict__ u_std,
    const float* __restrict__ y_mean,  const float* __restrict__ y_std,
    const unsigned int* __restrict__ wp,
    float* __restrict__ out) {
  __shared__ unsigned short xsb[16 * ASTR];  // rollout state x_t (bf16)
  __shared__ unsigned short gsb[16 * ASTR];  // hidden-1 scratch
  __shared__ unsigned short hsb[16 * ASTR];  // hidden-2 scratch

  int lane = threadIdx.x & 31;
  int b0   = blockIdx.x * 16;
  int brow = b0 + (lane & 15);
  int nl   = lane & 15;
  int kb2  = (lane & 16) ? 8 : 0;

  // ---- per-lane constants (registers) ----
  UN un, yn;
#pragma unroll
  for (int j = 0; j < 4; ++j) {
    float2 m = *(const float2*)(u_mean + kb2 + 2 * j);
    float2 s = *(const float2*)(u_std  + kb2 + 2 * j);
    un.m[j] = m; un.i[j].x = 1.f / s.x; un.i[j].y = 1.f / s.y;
    float2 my = *(const float2*)(y_mean + kb2 + 2 * j);
    float2 sy = *(const float2*)(y_std  + kb2 + 2 * j);
    yn.m[j] = my; yn.i[j].x = 1.f / sy.x; yn.i[j].y = 1.f / sy.y;
  }
  float fb1v[4], fb2v[4], fbxv[4], hb1v[4], hb2v[4];
#pragma unroll
  for (int n = 0; n < 4; ++n) {
    fb1v[n] = f_b1[n * 16 + nl];
    fb2v[n] = f_b2[n * 16 + nl];
    fbxv[n] = f_bres[n * 16 + nl] + f_b3[n * 16 + nl];
    hb1v[n] = h_b1[n * 16 + nl];
    hb2v[n] = h_b2[n * 16 + nl];
  }
  float hbyv = h_bres[nl] + h_b3[nl];
  float ysn  = y_std[nl];
  float ymn  = y_mean[nl];

  // ============================ encoder ============================
  {
    v8f a1[4], aR[4];
#pragma unroll
    for (int n = 0; n < 4; ++n) { a1[n] = vzero(); aR[n] = vzero(); }
    for (int kc = 0; kc < 32; ++kc) {       // uniform: no EXEC divergence
      int base = kc * 32;
      const float* src = (base < 512) ? (upast + (long)brow * 512)
                                      : (ypast + (long)brow * 512);
      const UN& nm = (base < 512) ? un : yn;
      if (base >= 512) base -= 512;
      v8u t;
#pragma unroll
      for (int r = 0; r < 8; ++r) {
        int k = base + kb2 + ((r & 4) << 2) + 2 * (r & 3);
        float2 v = *(const float2*)(src + k);
        int j = r & 3;
        t[r] = pack2bf((v.x - nm.m[j].x) * nm.i[j].x,
                       (v.y - nm.m[j].y) * nm.i[j].y);
      }
      v16bf a = __builtin_bit_cast(v16bf, t);
#pragma unroll
      for (int n = 0; n < 4; ++n) {
        a1[n] = wmma_bf16(a, b_frag(wp + OFF_ENC_W1,   n, kc, 32, lane), a1[n]);
        aR[n] = wmma_bf16(a, b_frag(wp + OFF_ENC_WRES, n, kc, 32, lane), aR[n]);
      }
    }
#pragma unroll
    for (int n = 0; n < 4; ++n)
      d_storeh<true>(gsb, n * 16, a1[n], enc_b1[n * 16 + nl], lane);

    v16bf g0 = a_ldsh(gsb, 0, lane), g1 = a_ldsh(gsb, 32, lane);
#pragma unroll
    for (int n = 0; n < 4; ++n) {
      v8f acc = vzero();
      acc = wmma_bf16(g0, b_frag(wp + OFF_ENC_W2, n, 0, 2, lane), acc);
      acc = wmma_bf16(g1, b_frag(wp + OFF_ENC_W2, n, 1, 2, lane), acc);
      d_storeh<true>(hsb, n * 16, acc, enc_b2[n * 16 + nl], lane);
    }
    v16bf h0 = a_ldsh(hsb, 0, lane), h1 = a_ldsh(hsb, 32, lane);
#pragma unroll
    for (int n = 0; n < 4; ++n) {
      v8f acc = aR[n];
      acc = wmma_bf16(h0, b_frag(wp + OFF_ENC_W3, n, 0, 2, lane), acc);
      acc = wmma_bf16(h1, b_frag(wp + OFF_ENC_W3, n, 1, 2, lane), acc);
      d_storeh<false>(xsb, n * 16, acc,
                      enc_bres[n * 16 + nl] + enc_b3[n * 16 + nl], lane);  // x0
    }
  }

  // ---- preload ALL rollout weights into registers (60 frags = 480 VGPRs) ----
  v16bf wF1[12], wFR[12], wF2[8], wF3[8], wH1[8], wH2[8], wHR[2], wH3[2];
#pragma unroll
  for (int n = 0; n < 4; ++n) {
#pragma unroll
    for (int c = 0; c < 3; ++c) {
      wF1[n * 3 + c] = b_frag(wp + OFF_F_W1,   n, c, 3, lane);
      wFR[n * 3 + c] = b_frag(wp + OFF_F_WRES, n, c, 3, lane);
    }
#pragma unroll
    for (int c = 0; c < 2; ++c) {
      wF2[n * 2 + c] = b_frag(wp + OFF_F_W2, n, c, 2, lane);
      wF3[n * 2 + c] = b_frag(wp + OFF_F_W3, n, c, 2, lane);
      wH1[n * 2 + c] = b_frag(wp + OFF_H_W1, n, c, 2, lane);
      wH2[n * 2 + c] = b_frag(wp + OFF_H_W2, n, c, 2, lane);
    }
  }
#pragma unroll
  for (int c = 0; c < 2; ++c) {
    wHR[c] = b_frag(wp + OFF_H_WRES, 0, c, 2, lane);
    wH3[c] = b_frag(wp + OFF_H_W3,   0, c, 2, lane);
  }

  // ============================ rollout ============================
  const float* urow = ufuture + ((long)brow * T_) * NU;
  int mb = (lane & 16) ? 8 : 0;
  // y output base for this lane: row (b0+mb), column nl; row stride T*NY
  float* ybase = out + ((long)(b0 + mb) * T_) * NY + nl;

  for (int t = 0; t < T_; ++t) {
    if (t + 1 < T_) __builtin_prefetch(urow + (t + 1) * NU, 0, 1);

    v16bf x0f = a_ldsh(xsb, 0, lane);           // x_t cols 0..31
    v16bf x1f = a_ldsh(xsb, 32, lane);          // x_t cols 32..63

    // fnet input chunk 2: u_n in cols 64..79, zeros 80..95 (r>=4 uniform zero)
    v16bf uf;
    {
      const float2* s2 = (const float2*)(urow + t * NU + kb2);
      v8u tt;
#pragma unroll
      for (int r = 0; r < 4; ++r) {
        float2 v = s2[r];
        tt[r] = pack2bf((v.x - un.m[r].x) * un.i[r].x,
                        (v.y - un.m[r].y) * un.i[r].y);
      }
#pragma unroll
      for (int r = 4; r < 8; ++r) tt[r] = 0u;
      uf = __builtin_bit_cast(v16bf, tt);
    }

    // ---- hnet(x_t) -> y[:, t, :]
#pragma unroll
    for (int n = 0; n < 4; ++n) {
      v8f acc = vzero();
      acc = wmma_bf16(x0f, wH1[n * 2 + 0], acc);
      acc = wmma_bf16(x1f, wH1[n * 2 + 1], acc);
      d_storeh<true>(gsb, n * 16, acc, hb1v[n], lane);
    }
    {
      v16bf g0 = a_ldsh(gsb, 0, lane), g1 = a_ldsh(gsb, 32, lane);
#pragma unroll
      for (int n = 0; n < 4; ++n) {
        v8f acc = vzero();
        acc = wmma_bf16(g0, wH2[n * 2 + 0], acc);
        acc = wmma_bf16(g1, wH2[n * 2 + 1], acc);
        d_storeh<true>(hsb, n * 16, acc, hb2v[n], lane);
      }
      v16bf h0 = a_ldsh(hsb, 0, lane), h1 = a_ldsh(hsb, 32, lane);
      v8f acc = vzero();
      acc = wmma_bf16(x0f, wHR[0], acc);
      acc = wmma_bf16(x1f, wHR[1], acc);
      acc = wmma_bf16(h0,  wH3[0], acc);
      acc = wmma_bf16(h1,  wH3[1], acc);
      // y = (acc + bres + b3) * y_std + y_mean
      float* yp = ybase + (long)t * NY;
#pragma unroll
      for (int v = 0; v < 8; ++v)
        yp[(long)v * T_ * NY] = (acc[v] + hbyv) * ysn + ymn;
    }

    // ---- fnet([x_t, u_t]) -> x_{t+1}
#pragma unroll
    for (int n = 0; n < 4; ++n) {
      v8f acc = vzero();
      acc = wmma_bf16(x0f, wF1[n * 3 + 0], acc);
      acc = wmma_bf16(x1f, wF1[n * 3 + 1], acc);
      acc = wmma_bf16(uf,  wF1[n * 3 + 2], acc);
      d_storeh<true>(gsb, n * 16, acc, fb1v[n], lane);
    }
    {
      v16bf g0 = a_ldsh(gsb, 0, lane), g1 = a_ldsh(gsb, 32, lane);
#pragma unroll
      for (int n = 0; n < 4; ++n) {
        v8f acc = vzero();
        acc = wmma_bf16(g0, wF2[n * 2 + 0], acc);
        acc = wmma_bf16(g1, wF2[n * 2 + 1], acc);
        d_storeh<true>(hsb, n * 16, acc, fb2v[n], lane);
      }
      v16bf h0 = a_ldsh(hsb, 0, lane), h1 = a_ldsh(hsb, 32, lane);
#pragma unroll
      for (int n = 0; n < 4; ++n) {
        v8f acc = vzero();
        acc = wmma_bf16(x0f, wFR[n * 3 + 0], acc);
        acc = wmma_bf16(x1f, wFR[n * 3 + 1], acc);
        acc = wmma_bf16(uf,  wFR[n * 3 + 2], acc);
        acc = wmma_bf16(h0,  wF3[n * 2 + 0], acc);
        acc = wmma_bf16(h1,  wF3[n * 2 + 1], acc);
        d_storeh<false>(xsb, n * 16, acc, fbxv[n], lane);   // x_{t+1}
      }
    }
  }
}

// ------------------------------ launcher -----------------------------------
// Input order (setup_inputs dict, nested dicts flattened in insertion order):
// 0 upast 1 ypast 2 ufuture
// 3..10  enc:  Wres bres W1 b1 W2 b2 W3 b3
// 11..18 fnet: Wres bres W1 b1 W2 b2 W3 b3
// 19..26 hnet: Wres bres W1 b1 W2 b2 W3 b3
// 27..30 norm: u_mean u_std y_mean y_std

extern "C" void kernel_launch(void* const* d_in, const int* in_sizes, int n_in,
                              void* d_out, int out_size, void* d_ws, size_t ws_size,
                              hipStream_t stream) {
  const float* upast    = (const float*)d_in[0];
  const float* ypast    = (const float*)d_in[1];
  const float* ufuture  = (const float*)d_in[2];
  const float* encWres  = (const float*)d_in[3];
  const float* enc_bres = (const float*)d_in[4];
  const float* encW1    = (const float*)d_in[5];
  const float* enc_b1   = (const float*)d_in[6];
  const float* encW2    = (const float*)d_in[7];
  const float* enc_b2   = (const float*)d_in[8];
  const float* encW3    = (const float*)d_in[9];
  const float* enc_b3   = (const float*)d_in[10];
  const float* fWres    = (const float*)d_in[11];
  const float* f_bres   = (const float*)d_in[12];
  const float* fW1      = (const float*)d_in[13];
  const float* f_b1     = (const float*)d_in[14];
  const float* fW2      = (const float*)d_in[15];
  const float* f_b2     = (const float*)d_in[16];
  const float* fW3      = (const float*)d_in[17];
  const float* f_b3     = (const float*)d_in[18];
  const float* hWres    = (const float*)d_in[19];
  const float* h_bres   = (const float*)d_in[20];
  const float* hW1      = (const float*)d_in[21];
  const float* h_b1     = (const float*)d_in[22];
  const float* hW2      = (const float*)d_in[23];
  const float* h_b2     = (const float*)d_in[24];
  const float* hW3      = (const float*)d_in[25];
  const float* h_b3     = (const float*)d_in[26];
  const float* u_mean   = (const float*)d_in[27];
  const float* u_std    = (const float*)d_in[28];
  const float* y_mean   = (const float*)d_in[29];
  const float* y_std    = (const float*)d_in[30];

  unsigned int* wp = (unsigned int*)d_ws;

  auto pack = [&](const float* W, int off, int K, int Kp, int N) {
    int total = (N >> 4) * (Kp >> 5) * 256;
    pack_w<<<(total + 255) / 256, 256, 0, stream>>>(W, wp + off, K, Kp, N);
  };
  pack(encW1,   OFF_ENC_W1,   1024, 1024, 64);
  pack(encWres, OFF_ENC_WRES, 1024, 1024, 64);
  pack(encW2,   OFF_ENC_W2,     64,   64, 64);
  pack(encW3,   OFF_ENC_W3,     64,   64, 64);
  pack(fW1,     OFF_F_W1,       80,   96, 64);
  pack(fWres,   OFF_F_WRES,     80,   96, 64);
  pack(fW2,     OFF_F_W2,       64,   64, 64);
  pack(fW3,     OFF_F_W3,       64,   64, 64);
  pack(hW1,     OFF_H_W1,       64,   64, 64);
  pack(hW2,     OFF_H_W2,       64,   64, 64);
  pack(hWres,   OFF_H_WRES,     64,   64, 16);
  pack(hW3,     OFF_H_W3,       64,   64, 16);

  dim3 grid(B_TOT / 16), block(32);   // 1 wave per workgroup, 16 batch rows per wave
  subnet_kernel<<<grid, block, 0, stream>>>(
      upast, ypast, ufuture,
      enc_bres, enc_b1, enc_b2, enc_b3,
      f_bres, f_b1, f_b2, f_b3,
      h_bres, h_b1, h_b2, h_b3,
      u_mean, u_std, y_mean, y_std,
      wp, (float*)d_out);
}
